// GraphEncoder_85804856639971
// MI455X (gfx1250) — compile-verified, matrix operand order checked by reading, not run on hardware
//
#include <hip/hip_runtime.h>

// ---------------------------------------------------------------------------
// CDNA5 (gfx1250) GCN encoder: WMMA f16 GEMMs with pre-packed B fragments,
// compile-time strides, software-pipelined loads, and per-wave B residency.
// ---------------------------------------------------------------------------

typedef __attribute__((ext_vector_type(16))) _Float16 v16h;
typedef __attribute__((ext_vector_type(8)))  float    v8f;

#define DEV __device__ __forceinline__

// A fragment: 16x32 (MxK) f16, row-major source with leading dim lda.
// ISA 7.12.2: lanes 0-15 hold row M=lane, K=0..7 & 16..23 as pairs in v0..v7;
// lanes 16-31 hold the same rows, K=8..15 & 24..31.
DEV v16h load_a_frag(const _Float16* A, int lda, int row0, int k0) {
  const int lane = threadIdx.x & 31;
  const int row  = row0 + (lane & 15);
  const int kh   = (lane >> 4) << 3;          // 0 or 8
  const _Float16* p = A + (size_t)row * lda + k0 + kh;
  v16h a;
#pragma unroll
  for (int v = 0; v < 8; ++v) {
    const int kb = ((v & 4) << 2) | ((v & 3) << 1);   // {0,2,4,6,16,18,20,22}
    a[2 * v]     = p[kb];
    a[2 * v + 1] = p[kb + 1];
  }
  return a;
}

DEV v8f wmma_f16(v16h a, v16h b, v8f c) {
  return __builtin_amdgcn_wmma_f32_16x16x32_f16(
      /*neg_a=*/false, a, /*neg_b=*/false, b,
      /*c_mod=*/(short)0, c, /*reuse_a=*/false, /*reuse_b=*/false);
}

// ---------------------------------------------------------------------------
// Weight pack: f32 B[K,srcN] row-major -> f16 WMMA B-fragment layout
// out[idx], idx = ((kt*(N/16))+tc)*512 + lane*16 + i ; zero-pads cols >= srcN.
// A B fragment (kt,tc) is then v16h[(kt*ntc+tc)*32 + lane]: one contiguous
// 32-byte per-lane load (2x global_load_b128), zero per-fragment addr math.
// ---------------------------------------------------------------------------
__global__ void k_pack_b(const float* __restrict__ B, _Float16* __restrict__ out,
                         int K, int N, int srcN) {
  int idx = blockIdx.x * blockDim.x + threadIdx.x;
  if (idx >= K * N) return;
  const int i    = idx & 15;
  const int lane = (idx >> 4) & 31;
  const int tile = idx >> 9;
  const int ntc  = N >> 4;
  const int tc   = tile % ntc;
  const int kt   = tile / ntc;
  const int col  = tc * 16 + (lane & 15);
  const int kh   = (lane >> 4) << 3;
  const int v    = i >> 1;
  const int kb   = (((v & 4) << 2) | ((v & 3) << 1)) + (i & 1);
  const int k    = kt * 32 + kh + kb;
  const float val = (col < srcN) ? B[(size_t)k * srcN + col] : 0.0f;
  out[idx] = (_Float16)val;
}

// ---------------------------------------------------------------------------
// Elementwise helpers
// ---------------------------------------------------------------------------
__global__ void k_cast_f16(const float* __restrict__ in, _Float16* __restrict__ out, int n) {
  int i = blockIdx.x * blockDim.x + threadIdx.x;
  if (i < n) out[i] = (_Float16)in[i];
}

__global__ void k_fill(float* __restrict__ p, float v, int n) {
  int i = blockIdx.x * blockDim.x + threadIdx.x;
  if (i < n) p[i] = v;
}

__global__ void k_deg_accum(const int* __restrict__ dst, float* __restrict__ deg, int e) {
  int i = blockIdx.x * blockDim.x + threadIdx.x;
  if (i < e) atomicAdd(&deg[dst[i]], 1.0f);
}

__global__ void k_rsqrt(const float* __restrict__ deg, float* __restrict__ dinv, int n) {
  int i = blockIdx.x * blockDim.x + threadIdx.x;
  if (i < n) {
    float d = deg[i];
    dinv[i] = (d > 0.0f) ? rsqrtf(d) : 0.0f;
  }
}

// out = h * dinv^2 (self-loop term initializes the aggregation buffer)
__global__ void k_agg_init(const float* __restrict__ h, const float* __restrict__ dinv,
                           float* __restrict__ out, int n, int F) {
  int i = blockIdx.x * blockDim.x + threadIdx.x;
  if (i < n) {
    float dv = dinv[i / F];
    out[i] = h[i] * dv * dv;
  }
}

// out[dst] += h[src] * dinv[src] * dinv[dst], 4 features per thread
__global__ void k_agg_edges4(const float4* __restrict__ h4, const float* __restrict__ dinv,
                             const int* __restrict__ src, const int* __restrict__ dst,
                             float* __restrict__ out, int totalq, int Fq) {
  int i = blockIdx.x * blockDim.x + threadIdx.x;
  if (i >= totalq) return;
  int e = i / Fq;
  int q = i - e * Fq;
  int s = src[e], d = dst[e];
  float wgt = dinv[s] * dinv[d];
  float4 v = h4[(size_t)s * Fq + q];
  float* op = out + ((size_t)d * Fq + q) * 4;
  atomicAdd(op + 0, v.x * wgt);
  atomicAdd(op + 1, v.y * wgt);
  atomicAdd(op + 2, v.z * wgt);
  atomicAdd(op + 3, v.w * wgt);
}

// h16 = f16(relu(agg + bias))
__global__ void k_bias_relu_f16(const float* __restrict__ in, const float* __restrict__ bias,
                                _Float16* __restrict__ out, int n, int F) {
  int i = blockIdx.x * blockDim.x + threadIdx.x;
  if (i < n) {
    float v = in[i] + bias[i % F];
    out[i] = (_Float16)(v > 0.0f ? v : 0.0f);
  }
}

// f(out, f32) = agg + bias ; f16 copy for downstream GEMMs
__global__ void k_finalize_f(const float* __restrict__ in, const float* __restrict__ bias,
                             float* __restrict__ outf, _Float16* __restrict__ outh, int n, int F) {
  int i = blockIdx.x * blockDim.x + threadIdx.x;
  if (i < n) {
    float v = in[i] + bias[i % F];
    outf[i] = v;
    outh[i] = (_Float16)v;
  }
}

__global__ void k_copy_ei(const int* __restrict__ ei, float* __restrict__ out, int n) {
  int i = blockIdx.x * blockDim.x + threadIdx.x;
  if (i < n) out[i] = (float)ei[i];
}

// ---------------------------------------------------------------------------
// WMMA GEMM (compile-time K,N): C[M,N] f32 = A[M,K] f16 @ Bpacked f16
// One 16x64 strip per wave, 4 accumulators per A fragment, depth-1 software
// pipeline (next K-step's fragments load under current WMMAs). All fragment
// addresses are base + immediate offset. Strip id is wave-uniform -> EXEC=~0.
// ---------------------------------------------------------------------------
template <int K, int N>
__global__ __launch_bounds__(128) void k_gemm_f16(const _Float16* __restrict__ A,
                                                  const _Float16* __restrict__ Bp,
                                                  float* __restrict__ C, int M) {
  constexpr int KT = K / 32;          // K-steps
  constexpr int NTC = N / 16;         // 16-col tiles
  constexpr int NSC = N / 64;         // 64-col strips
  constexpr int BSTEP = NTC * 32;     // v16h per K-step
  const int strips = (M >> 4) * NSC;
  const int t = blockIdx.x * 4 + (threadIdx.x >> 5);
  if (t >= strips) return;
  const int tr = t / NSC, sc = t % NSC;
  const int lane = threadIdx.x & 31;
  const v16h* bp = (const v16h*)Bp + (size_t)sc * 128 + lane;

  v8f acc[4] = {{}, {}, {}, {}};
  v16h a  = load_a_frag(A, K, tr * 16, 0);
  v16h b0 = bp[0], b1 = bp[32], b2 = bp[64], b3 = bp[96];
#pragma unroll
  for (int kt = 0; kt < KT; ++kt) {
    v16h an, c0, c1, c2, c3;
    if (kt + 1 < KT) {
      const v16h* bn = bp + (size_t)(kt + 1) * BSTEP;
      an = load_a_frag(A, K, tr * 16, (kt + 1) * 32);
      c0 = bn[0]; c1 = bn[32]; c2 = bn[64]; c3 = bn[96];
    }
    acc[0] = wmma_f16(a, b0, acc[0]);
    acc[1] = wmma_f16(a, b1, acc[1]);
    acc[2] = wmma_f16(a, b2, acc[2]);
    acc[3] = wmma_f16(a, b3, acc[3]);
    if (kt + 1 < KT) { a = an; b0 = c0; b1 = c1; b2 = c2; b3 = c3; }
  }
  const int mbase = tr * 16 + ((lane >> 4) << 3);
#pragma unroll
  for (int j = 0; j < 4; ++j) {
    const int n = (sc * 4 + j) * 16 + (lane & 15);
#pragma unroll
    for (int r = 0; r < 8; ++r)
      C[(size_t)(mbase + r) * N + n] = acc[j][r];
  }
}

// ---------------------------------------------------------------------------
// Fused edge MLP: out[e] = relu(concat(f[src],f[dst]) @ Wp1 + bp1) @ Wp2 + bp2
// 64 edges per 256-thread block (8 wave32s). ef and hidden staged in LDS with
// padded leading dims; B fragments hoisted to registers (loaded once per wave).
// ---------------------------------------------------------------------------
#define EPB    64
#define EF_LD  136   // 128 + 8 pad (272 B row: 16B-multiple)
#define HID_LD 264   // 256 + 8 pad

__global__ __launch_bounds__(256) void k_edge_mlp(
    const _Float16* __restrict__ f16v,   // [N,64] f16
    const int* __restrict__ src, const int* __restrict__ dst,
    const _Float16* __restrict__ Wp1p,   // packed [128,256]
    const float* __restrict__ bp1,       // [256]
    const _Float16* __restrict__ Wp2p,   // packed [256,64]
    const float* __restrict__ bp2,       // [64]
    float* __restrict__ out) {           // [E,64]
  __shared__ alignas(16) _Float16 sEF[EPB * EF_LD];
  __shared__ alignas(16) _Float16 sHID[EPB * HID_LD];
  __shared__ int sIdx[2 * EPB];

  const int tid = threadIdx.x;
  const int e0 = blockIdx.x * EPB;

  if (tid < EPB)          sIdx[tid] = src[e0 + tid];
  else if (tid < 2 * EPB) sIdx[tid] = dst[e0 + tid - EPB];
  __syncthreads();

  // Gather ef rows in 16B chunks: per edge, chunks 0..7 = f[src], 8..15 = f[dst]
  const float4* fv4 = (const float4*)f16v;      // node row = 8 x float4
#pragma unroll
  for (int i = tid; i < EPB * 16; i += 256) {
    int e = i >> 4;
    int q = i & 15;
    int node = (q < 8) ? sIdx[e] : sIdx[EPB + e];
    *(float4*)(&sEF[e * EF_LD + q * 8]) = fv4[(size_t)node * 8 + (q & 7)];
  }

  const int wave = tid >> 5;
  const int lane = tid & 31;

  // Hoist GEMM1 B fragments: depend only on (kt, tc); tc = wave, wave+8.
  // Loaded once per wave (64 VGPRs) and reused across all 4 row tiles.
  const v16h* b1base = (const v16h*)Wp1p + lane;   // frag(kt,tc) at +tc*32+kt*512
  v16h B1[2][4];
  float bn1[2];
#pragma unroll
  for (int j = 0; j < 2; ++j) {
    const int tc = wave + (j << 3);
    const v16h* bp = b1base + tc * 32;
    B1[j][0] = bp[0];
    B1[j][1] = bp[512];
    B1[j][2] = bp[1024];
    B1[j][3] = bp[1536];
    bn1[j] = bp1[tc * 16 + (lane & 15)];
  }
  __syncthreads();

  // GEMM1: hid[64,256] = relu(sEF[64,128] @ Wp1 + bp1), stored f16 in LDS.
  // Each wave: all 4 row tiles x 2 column tiles; A fragments shared across tc.
#pragma unroll
  for (int tr = 0; tr < 4; ++tr) {
    v16h a0 = load_a_frag(sEF, EF_LD, tr * 16, 0);
    v16h a1 = load_a_frag(sEF, EF_LD, tr * 16, 32);
    v16h a2 = load_a_frag(sEF, EF_LD, tr * 16, 64);
    v16h a3 = load_a_frag(sEF, EF_LD, tr * 16, 96);
    const int mbase = tr * 16 + ((lane >> 4) << 3);
#pragma unroll
    for (int j = 0; j < 2; ++j) {
      const int tc = wave + (j << 3);
      v8f acc = {};
      acc = wmma_f16(a0, B1[j][0], acc);
      acc = wmma_f16(a1, B1[j][1], acc);
      acc = wmma_f16(a2, B1[j][2], acc);
      acc = wmma_f16(a3, B1[j][3], acc);
      const int n = tc * 16 + (lane & 15);
#pragma unroll
      for (int r = 0; r < 8; ++r) {
        float v = acc[r] + bn1[j];
        sHID[(mbase + r) * HID_LD + n] = (_Float16)(v > 0.0f ? v : 0.0f);
      }
    }
  }

  // Hoist GEMM2 B fragments: both of this wave's tiles share tc = wave & 3,
  // so the 8 fragments (64 VGPRs) are loaded exactly once.
  const int tc2 = wave & 3;
  const int trb = wave >> 2;                       // 0 or 1
  const v16h* bp2v = (const v16h*)Wp2p + tc2 * 32 + lane;  // +kt*128
  v16h B2[8];
#pragma unroll
  for (int kt = 0; kt < 8; ++kt) B2[kt] = bp2v[kt * 128];
  const float bn2 = bp2[tc2 * 16 + (lane & 15)];
  __syncthreads();

  // GEMM2: out[64,64] = sHID[64,256] @ Wp2 + bp2; wave does tr = trb, trb+2.
#pragma unroll
  for (int ti = 0; ti < 2; ++ti) {
    const int tr = trb + ti * 2;
    v8f acc = {};
#pragma unroll
    for (int kt = 0; kt < 8; ++kt) {
      v16h a = load_a_frag(sHID, HID_LD, tr * 16, kt * 32);
      acc = wmma_f16(a, B2[kt], acc);
    }
    const int n = tc2 * 16 + (lane & 15);
    const int mbase = tr * 16 + ((lane >> 4) << 3);
#pragma unroll
    for (int r = 0; r < 8; ++r)
      out[(size_t)(e0 + mbase + r) * 64 + n] = acc[r] + bn2;
  }
}

// ---------------------------------------------------------------------------
// Logits: f[N,64] @ WcPacked[64,48] + bc, masked store to [N,40]
// ---------------------------------------------------------------------------
__global__ __launch_bounds__(128) void k_logits(const _Float16* __restrict__ f16v,
                                                const _Float16* __restrict__ Wcp,
                                                const float* __restrict__ bc,
                                                float* __restrict__ out, int Nn) {
  const int ntc = 3;                       // 48/16 column tiles
  const int tiles = (Nn >> 4) * ntc;
  const int t = blockIdx.x * 4 + (threadIdx.x >> 5);
  if (t >= tiles) return;
  const int tr = t / ntc, tc = t % ntc;
  const int lane = threadIdx.x & 31;
  const v16h* bp = (const v16h*)Wcp + tc * 32 + lane;  // frag(kt,tc): +kt*96
  v8f acc = {};
  {
    v16h a0 = load_a_frag(f16v, 64, tr * 16, 0);
    v16h a1 = load_a_frag(f16v, 64, tr * 16, 32);
    acc = wmma_f16(a0, bp[0], acc);
    acc = wmma_f16(a1, bp[96], acc);
  }
  const int n = tc * 16 + (lane & 15);
  const int mbase = tr * 16 + ((lane >> 4) << 3);
  if (n < 40) {
    const float bn = bc[n];
#pragma unroll
    for (int r = 0; r < 8; ++r)
      out[(size_t)(mbase + r) * 40 + n] = acc[r] + bn;
  }
}

// ---------------------------------------------------------------------------
// Launcher
// ---------------------------------------------------------------------------
extern "C" void kernel_launch(void* const* d_in, const int* in_sizes, int n_in,
                              void* d_out, int out_size, void* d_ws, size_t ws_size,
                              hipStream_t stream) {
  (void)in_sizes; (void)n_in; (void)out_size; (void)ws_size;
  constexpr int N = 50000, E = 800000, IN = 128, HID = 256, OUT = 64, NC = 40, NCP = 48;

  const float* x   = (const float*)d_in[0];
  const int*   ei  = (const int*)  d_in[1];
  const float* W1  = (const float*)d_in[2];
  const float* b1  = (const float*)d_in[3];
  const float* W2  = (const float*)d_in[4];
  const float* b2  = (const float*)d_in[5];
  const float* Wp1 = (const float*)d_in[6];
  const float* bp1 = (const float*)d_in[7];
  const float* Wp2 = (const float*)d_in[8];
  const float* bp2 = (const float*)d_in[9];
  const float* Wc  = (const float*)d_in[10];
  const float* bc  = (const float*)d_in[11];
  const int* src = ei;
  const int* dst = ei + E;

  float* out_f  = (float*)d_out;                         // [N,64]
  float* out_ef = out_f  + (size_t)N * OUT;              // [E,64]
  float* out_lg = out_ef + (size_t)E * OUT;              // [N,40]
  float* out_ei = out_lg + (size_t)N * NC;               // [2,E] as float

  // Workspace layout (256B-aligned slabs, with time-disjoint reuse)
  char* w = (char*)d_ws;
  size_t o = 0;
  auto take = [&](size_t bytes) -> char* {
    char* p = w + o;
    o += (bytes + 255) & ~(size_t)255;
    return p;
  };
  _Float16* x_h   = (_Float16*)take((size_t)N * IN * 2);    // reused as f_h
  _Float16* W1_p  = (_Float16*)take((size_t)IN * HID * 2);
  _Float16* W2_p  = (_Float16*)take((size_t)HID * OUT * 2);
  _Float16* Wp1_p = (_Float16*)take((size_t)128 * 256 * 2);
  _Float16* Wp2_p = (_Float16*)take((size_t)256 * 64 * 2);
  _Float16* Wc_p  = (_Float16*)take((size_t)OUT * NCP * 2);
  float*    deg   = (float*)take((size_t)N * 4);
  float*    dinv  = (float*)take((size_t)N * 4);
  float*    h0    = (float*)take((size_t)N * HID * 4);      // reused as f0
  float*    agg1  = (float*)take((size_t)N * HID * 4);      // reused as agg2
  _Float16* h_h   = (_Float16*)take((size_t)N * HID * 2);
  _Float16* f_h   = x_h;     // x_h dead after GEMM1
  float*    f0    = h0;      // h0 dead after layer-1 aggregation
  float*    agg2  = agg1;    // agg1 dead after finalize1

  auto cdiv = [](long long a, long long b) { return (unsigned)((a + b - 1) / b); };

  // --- activation cast + weight packing (f32 -> f16 WMMA B-fragment layout) ---
  k_cast_f16<<<cdiv((long long)N * IN, 256), 256, 0, stream>>>(x, x_h, N * IN);
  k_pack_b<<<cdiv(IN * HID, 256), 256, 0, stream>>>(W1, W1_p, IN, HID, HID);
  k_pack_b<<<cdiv(HID * OUT, 256), 256, 0, stream>>>(W2, W2_p, HID, OUT, OUT);
  k_pack_b<<<cdiv(128 * 256, 256), 256, 0, stream>>>(Wp1, Wp1_p, 128, 256, 256);
  k_pack_b<<<cdiv(256 * 64, 256), 256, 0, stream>>>(Wp2, Wp2_p, 256, 64, 64);
  k_pack_b<<<cdiv(OUT * NCP, 256), 256, 0, stream>>>(Wc, Wc_p, OUT, NCP, NC);

  // --- symmetric normalization: deg = in-degree + 1 (self loop), dinv = rsqrt ---
  k_fill<<<cdiv(N, 256), 256, 0, stream>>>(deg, 1.0f, N);
  k_deg_accum<<<cdiv(E, 256), 256, 0, stream>>>(dst, deg, E);
  k_rsqrt<<<cdiv(N, 256), 256, 0, stream>>>(deg, dinv, N);

  // --- layer 1: h = relu(Dinv (A+I) Dinv (x @ W1) + b1) ---
  k_gemm_f16<IN, HID><<<cdiv((N / 16) * (HID / 64), 4), 128, 0, stream>>>(x_h, W1_p, h0, N);
  k_agg_init<<<cdiv((long long)N * HID, 256), 256, 0, stream>>>(h0, dinv, agg1, N * HID, HID);
  k_agg_edges4<<<cdiv((long long)E * (HID / 4), 256), 256, 0, stream>>>(
      (const float4*)h0, dinv, src, dst, agg1, E * (HID / 4), HID / 4);
  k_bias_relu_f16<<<cdiv((long long)N * HID, 256), 256, 0, stream>>>(agg1, b1, h_h, N * HID, HID);

  // --- layer 2: f = Dinv (A+I) Dinv (h @ W2) + b2 ---
  k_gemm_f16<HID, OUT><<<cdiv((N / 16) * (OUT / 64), 4), 128, 0, stream>>>(h_h, W2_p, f0, N);
  k_agg_init<<<cdiv((long long)N * OUT, 256), 256, 0, stream>>>(f0, dinv, agg2, N * OUT, OUT);
  k_agg_edges4<<<cdiv((long long)E * (OUT / 4), 256), 256, 0, stream>>>(
      (const float4*)f0, dinv, src, dst, agg2, E * (OUT / 4), OUT / 4);
  k_finalize_f<<<cdiv((long long)N * OUT, 256), 256, 0, stream>>>(agg2, b2, out_f, f_h,
                                                                  N * OUT, OUT);

  // --- edge MLP (dominant: ~79 GFLOP) ---
  k_edge_mlp<<<E / EPB, 256, 0, stream>>>(f_h, src, dst, Wp1_p, bp1, Wp2_p, bp2, out_ef);

  // --- logits ---
  k_logits<<<cdiv((N / 16) * 3, 4), 128, 0, stream>>>(f_h, Wc_p, bc, out_lg, N);

  // --- edge_index passthrough ---
  k_copy_ei<<<cdiv(2 * E, 256), 256, 0, stream>>>(ei, out_ei, 2 * E);
}
